// ERIC_33277406610158
// MI455X (gfx1250) — matrix operand order repaired and checked by reading, not compiled, not run on hardware
//
#include <hip/hip_runtime.h>
#include <hip/hip_bf16.h>
#include <math.h>

// Problem constants (from reference)
#define Lc 3
#define Dc 64
#define Bc 1024
#define Nc 131072
#define Ec 1048576
#define Tc 16
#define NDc ((size_t)Nc * Dc) /* 8388608 */

typedef _Float16 h16;
typedef __attribute__((ext_vector_type(16))) _Float16 v16h;
typedef __attribute__((ext_vector_type(8)))  float    v8f;
typedef __attribute__((ext_vector_type(4)))  unsigned int u32x4;

union Frag16 { v16h v; u32x4 q[2]; };

// ---------------------------------------------------------------------------
// WMMA GEMM: out[M x 64] = act(A[M x 64] @ W[64 x 64] + bias)
// A, W are f16 row-major. W staged through LDS once per workgroup.
// One wave = 16 rows x 64 cols; 256-thread block = 128 rows.
// Output paths selected at compile time: WF -> f32 (strided, col-offset),
// WH -> f16 (stride 64).
// ---------------------------------------------------------------------------
template<bool RELU, bool WF, bool WH>
__global__ __launch_bounds__(256) void gemm64_wmma(
    const h16* __restrict__ A, const h16* __restrict__ W,
    const float* __restrict__ bias,
    float* __restrict__ outF, h16* __restrict__ outH,
    int outStride, int outColOff)
{
    __shared__ __align__(16) h16 sW[64 * 64];   // 8 KB of the 320 KB WGP LDS

    // Cooperative coalesced stage of W into LDS: 256 threads x 32 B.
    {
        u32x4*       d = (u32x4*)sW;
        const u32x4* s = (const u32x4*)W;
        d[threadIdx.x * 2 + 0] = s[threadIdx.x * 2 + 0];
        d[threadIdx.x * 2 + 1] = s[threadIdx.x * 2 + 1];
    }
    __syncthreads();

    const int wave = threadIdx.x >> 5;
    const int lane = threadIdx.x & 31;
    const int hi   = lane >> 4;   // half-wave select
    const int ln   = lane & 15;
    const long m0  = (long)blockIdx.x * 128 + wave * 16;

    // A fragments for k-chunks {0,32}: 16-bit A 16x32 layout (ISA 7.12.2)
    // lane<16: K = kc+{0..7, 16..23}; lane>=16: K = kc+{8..15, 24..31}; row = ln
    Frag16 a[2];
    const h16* arow = A + (m0 + ln) * 64;
#pragma unroll
    for (int kc = 0; kc < 2; ++kc) {
        a[kc].q[0] = *(const u32x4*)(arow + kc * 32 + hi * 8);
        a[kc].q[1] = *(const u32x4*)(arow + kc * 32 + 16 + hi * 8);
    }

    // B fragments from LDS: lane holds N-column (nt*16 + ln), K = kc+{0..7,16..23} (lane<16) / kc+{8..15,24..31} (lane>=16).
    Frag16 b[4][2];
#pragma unroll
    for (int nt = 0; nt < 4; ++nt)
#pragma unroll
        for (int kc = 0; kc < 2; ++kc) {
            const h16* wcol = sW + (kc * 32) * 64 + nt * 16 + ln;
#pragma unroll
            for (int s = 0; s < 16; ++s) {
                const int k = (s < 8) ? (hi * 8 + s) : (16 + hi * 8 + (s - 8));
                b[nt][kc].v[s] = wcol[k * 64];
            }
        }

#pragma unroll
    for (int nt = 0; nt < 4; ++nt) {
        v8f c = {};
        c = __builtin_amdgcn_wmma_f32_16x16x32_f16(false, a[0].v, false, b[nt][0].v,
                                                   (short)0, c, false, false);
        c = __builtin_amdgcn_wmma_f32_16x16x32_f16(false, a[1].v, false, b[nt][1].v,
                                                   (short)0, c, false, false);
        // C/D layout: VGPR r -> row m0 + hi*8 + r, lane&15 -> column nt*16+ln
        const int  col   = nt * 16 + ln;
        const float bv   = bias[col];
        const long rbase = m0 + hi * 8;
        float*     pF    = WF ? (outF + rbase * (long)outStride + outColOff + col) : nullptr;
        h16*       pH    = WH ? (outH + rbase * 64 + col) : nullptr;
#pragma unroll
        for (int r = 0; r < 8; ++r) {
            float v = c[r] + bv;
            if (RELU) v = v > 0.f ? v : 0.f;
            if (WF) { *pF = v;      pF += outStride; }
            if (WH) { *pH = (h16)v; pH += 64;        }
        }
    }
}

// ---------------------------------------------------------------------------
// Elementwise / scatter / reduction helpers
// ---------------------------------------------------------------------------
__global__ void cvt_f32_f16_kernel(const float* __restrict__ s, h16* __restrict__ d, int n) {
    int i = blockIdx.x * blockDim.x + threadIdx.x;
    if (i < n) d[i] = (h16)s[i];
}

// agg[dst] += x[src] over 1M edges; 16 threads/edge, float4 gather + 4 atomics
__global__ void edge_scatter_kernel(const float* __restrict__ x, const int* __restrict__ ei,
                                    float* __restrict__ agg, int E) {
    long tid = (long)blockIdx.x * blockDim.x + threadIdx.x;
    int e = (int)(tid >> 4), q = (int)(tid & 15);
    if (e >= E) return;
    int src = ei[e], dst = ei[E + e];
    float4 v = ((const float4*)(x + (size_t)src * 64))[q];
    float* a = agg + (size_t)dst * 64 + q * 4;
    atomicAdd(a + 0, v.x); atomicAdd(a + 1, v.y);
    atomicAdd(a + 2, v.z); atomicAdd(a + 3, v.w);
}

__global__ void gin_pre_kernel(const float* __restrict__ x, const float* __restrict__ agg,
                               const float* __restrict__ eps, int layer,
                               h16* __restrict__ t16, size_t total) {
    size_t i = (size_t)blockIdx.x * blockDim.x + threadIdx.x;
    if (i >= total) return;
    float e1 = 1.0f + eps[layer];
    t16[i] = (h16)(e1 * x[i] + agg[i]);
}

// Per-column sum / sum-of-squares over all N rows (64 cols). 64 threads/block.
__global__ void bn_stats_kernel(const float* __restrict__ h, float* __restrict__ stats,
                                int rowsPerBlock) {
    int col = threadIdx.x;
    long base = (long)blockIdx.x * rowsPerBlock;
    float s = 0.f, s2 = 0.f;
    for (int r = 0; r < rowsPerBlock; ++r) {
        float v = h[(base + r) * 64 + col];
        s += v; s2 += v * v;
    }
    atomicAdd(&stats[col], s);
    atomicAdd(&stats[64 + col], s2);
}

__global__ void bn_apply_kernel(const float* __restrict__ h, const float* __restrict__ stats,
                                const float* __restrict__ gamma, const float* __restrict__ beta,
                                float* __restrict__ xf, h16* __restrict__ x16, size_t total) {
    size_t i = (size_t)blockIdx.x * blockDim.x + threadIdx.x;
    if (i >= total) return;
    int col = (int)(i & 63);
    const float invN = 1.0f / (float)Nc;
    float mu  = stats[col] * invN;
    float var = stats[64 + col] * invN - mu * mu;
    float v = (h[i] - mu) * rsqrtf(var + 1e-5f) * gamma[col] + beta[col];
    v = fmaxf(v, 0.f);
    xf[i] = v;
    x16[i] = (h16)v;
}

// pooled[gidx[n]][d] += inner[n][d]
__global__ void pool_seg_kernel(const float* __restrict__ inner, const int* __restrict__ gidx,
                                float* __restrict__ pooled, size_t total) {
    size_t i = (size_t)blockIdx.x * blockDim.x + threadIdx.x;
    if (i >= total) return;
    int n = (int)(i >> 6), d = (int)(i & 63);
    atomicAdd(&pooled[(size_t)gidx[n] * 64 + d], inner[i]);
}

// ---------------------------------------------------------------------------
// Head (B = 1024 rows; VALU is plenty)
// ---------------------------------------------------------------------------
__global__ void diff_kernel(const float* __restrict__ qf, const float* __restrict__ cf,
                            float* __restrict__ out, int n) {
    int i = blockIdx.x * blockDim.x + threadIdx.x;
    if (i >= n) return;
    float d = qf[i] - cf[i];
    out[i] = expf(-d * d);
}

// out[b,j] = act(sum_k in[b,k] * W[k,j] + bias[j]); act 0=relu 1=tanh
__global__ void mlp_kernel(const float* __restrict__ in, const float* __restrict__ W,
                           const float* __restrict__ bias, float* __restrict__ out,
                           int nb, int K, int J, int act) {
    int i = blockIdx.x * blockDim.x + threadIdx.x;
    if (i >= nb * J) return;
    int b = i / J, j = i % J;
    const float* row = in + (long)b * K;
    float s = bias[j];
    for (int k = 0; k < K; ++k) s += row[k] * W[(long)k * J + j];
    if (act == 0) s = fmaxf(s, 0.f);
    else          s = tanhf(s);
    out[i] = s;
}

__global__ void score_head_kernel(const float* __restrict__ h, const float* __restrict__ w1,
                                  const float* __restrict__ b1, const float* __restrict__ w2,
                                  const float* __restrict__ b2, float* __restrict__ score, int nb) {
    int b = blockIdx.x * blockDim.x + threadIdx.x;
    if (b >= nb) return;
    float t[16];
    for (int j = 0; j < 16; ++j) {
        float s = b1[j];
        for (int k = 0; k < 96; ++k) s += h[(long)b * 96 + k] * w1[k * 16 + j];
        t[j] = fmaxf(s, 0.f);
    }
    float s = b2[0];
    for (int j = 0; j < 16; ++j) s += t[j] * w2[j];
    score[b] = 1.f / (1.f + expf(-s));
}

// simrep[b,t] = relu( e1 W[:,:,t] e2 + [e1;e2].wb[t] + bias[t] )
__global__ void ntn_kernel(const float* __restrict__ qf, const float* __restrict__ cf,
                           const float* __restrict__ W, const float* __restrict__ wb,
                           const float* __restrict__ nbias, float* __restrict__ simrep, int nb) {
    int i = blockIdx.x * blockDim.x + threadIdx.x;
    if (i >= nb * 16) return;
    int b = i >> 4, t = i & 15;
    const float* e1 = qf + (long)b * 192 + 128;
    const float* e2 = cf + (long)b * 192 + 128;
    float acc = 0.f;
    for (int g = 0; g < 64; ++g) {
        float s = 0.f;
        for (int f = 0; f < 64; ++f) s += e1[f] * W[((long)f * 64 + g) * 16 + t];
        acc += s * e2[g];
    }
    float blk = 0.f;
    for (int f = 0; f < 64; ++f)
        blk += e1[f] * wb[t * 128 + f] + e2[f] * wb[t * 128 + 64 + f];
    simrep[i] = fmaxf(acc + blk + nbias[t], 0.f);
}

__global__ void sim_head_kernel(const float* __restrict__ rep, const float* __restrict__ w1,
                                const float* __restrict__ b1, const float* __restrict__ w2,
                                const float* __restrict__ b2, float* __restrict__ sim, int nb) {
    int b = blockIdx.x * blockDim.x + threadIdx.x;
    if (b >= nb) return;
    float t[16];
    for (int j = 0; j < 16; ++j) {
        float s = b1[j];
        for (int k = 0; k < 16; ++k) s += rep[b * 16 + k] * w1[k * 16 + j];
        t[j] = fmaxf(s, 0.f);
    }
    float s = b2[0];
    for (int j = 0; j < 16; ++j) s += t[j] * w2[j];
    sim[b] = 1.f / (1.f + expf(-s));
}

__global__ void final_combine_kernel(const float* __restrict__ score, const float* __restrict__ sim,
                                     const float* __restrict__ alpha, const float* __restrict__ beta,
                                     float* __restrict__ out, int nb) {
    int b = blockIdx.x * blockDim.x + threadIdx.x;
    if (b >= nb) return;
    out[b] = alpha[0] * score[b] + beta[0] * sim[b];
}

// ---------------------------------------------------------------------------
// Orchestration
// ---------------------------------------------------------------------------
extern "C" void kernel_launch(void* const* d_in, const int* in_sizes, int n_in,
                              void* d_out, int out_size, void* d_ws, size_t ws_size,
                              hipStream_t stream) {
    (void)in_sizes; (void)n_in; (void)out_size; (void)ws_size;

    const float* query_x  = (const float*)d_in[0];
    const float* corpus_x = (const float*)d_in[1];
    const int*   qei      = (const int*)d_in[2];
    const int*   cei      = (const int*)d_in[3];
    const int*   qgi      = (const int*)d_in[4];
    const int*   cgi      = (const int*)d_in[5];
    const float* gin_w1   = (const float*)d_in[6];
    const float* gin_b1   = (const float*)d_in[7];
    const float* gin_w2   = (const float*)d_in[8];
    const float* gin_b2   = (const float*)d_in[9];
    const float* bn_gamma = (const float*)d_in[10];
    const float* bn_beta  = (const float*)d_in[11];
    const float* gin_eps  = (const float*)d_in[12];
    const float* inner_w  = (const float*)d_in[13];
    const float* inner_b  = (const float*)d_in[14];
    const float* outer_w  = (const float*)d_in[15];
    const float* outer_b  = (const float*)d_in[16];
    const float* conv_w1  = (const float*)d_in[17];
    const float* conv_b1  = (const float*)d_in[18];
    const float* conv_w2  = (const float*)d_in[19];
    const float* conv_b2  = (const float*)d_in[20];
    const float* score_w1 = (const float*)d_in[21];
    const float* score_b1 = (const float*)d_in[22];
    const float* score_w2 = (const float*)d_in[23];
    const float* score_b2 = (const float*)d_in[24];
    const float* ntn_w    = (const float*)d_in[25];
    const float* ntn_wb   = (const float*)d_in[26];
    const float* ntn_bias = (const float*)d_in[27];
    const float* sim_w1   = (const float*)d_in[28];
    const float* sim_b1   = (const float*)d_in[29];
    const float* sim_w2   = (const float*)d_in[30];
    const float* sim_b2   = (const float*)d_in[31];
    const float* alpha    = (const float*)d_in[32];
    const float* beta     = (const float*)d_in[33];

    // ---- workspace layout (aliased) ----
    char* wp = (char*)d_ws;
    auto alloc = [&](size_t bytes) -> void* {
        void* p = (void*)wp;
        wp += (bytes + 255) & ~(size_t)255;
        return p;
    };
    float* xf   = (float*)alloc(NDc * 4);      // current node features (f32)
    float* buf1 = (float*)alloc(NDc * 4);      // agg, later h (pre-BN)
    float* buf2 = (float*)alloc(NDc * 4);      // [t16 | u16] halves, later inner f32
    h16*   t16  = (h16*)buf2;
    h16*   u16  = t16 + NDc;
    float* inner = buf2;
    h16*   x16  = (h16*)alloc(NDc * 2);        // post-BN features in f16
    float* stats   = (float*)alloc(128 * 4);
    float* pooled  = (float*)alloc((size_t)Bc * 64 * 4);
    h16*   pooled16 = (h16*)alloc((size_t)Bc * 64 * 2);
    h16*   w16g1 = (h16*)alloc((size_t)Lc * 4096 * 2);
    h16*   w16g2 = (h16*)alloc((size_t)Lc * 4096 * 2);
    h16*   w16in = (h16*)alloc((size_t)Lc * 4096 * 2);
    h16*   w16ot = (h16*)alloc((size_t)Lc * 4096 * 2);
    float* qf   = (float*)alloc((size_t)Bc * 192 * 4);
    float* cf   = (float*)alloc((size_t)Bc * 192 * 4);
    float* dbuf = (float*)alloc((size_t)Bc * 192 * 4);
    float* hc1  = (float*)alloc((size_t)Bc * 96 * 4);
    float* hc2  = (float*)alloc((size_t)Bc * 96 * 4);
    float* scoreb = (float*)alloc((size_t)Bc * 4);
    float* simrep = (float*)alloc((size_t)Bc * 16 * 4);
    float* simb   = (float*)alloc((size_t)Bc * 4);

    const int TB = 256;
    // ---- one-time weight conversion to f16 (cheap, rerun each call) ----
    cvt_f32_f16_kernel<<<(Lc * 4096 + TB - 1) / TB, TB, 0, stream>>>(gin_w1, w16g1, Lc * 4096);
    cvt_f32_f16_kernel<<<(Lc * 4096 + TB - 1) / TB, TB, 0, stream>>>(gin_w2, w16g2, Lc * 4096);
    cvt_f32_f16_kernel<<<(Lc * 4096 + TB - 1) / TB, TB, 0, stream>>>(inner_w, w16in, Lc * 4096);
    cvt_f32_f16_kernel<<<(Lc * 4096 + TB - 1) / TB, TB, 0, stream>>>(outer_w, w16ot, Lc * 4096);

    auto encode = [&](const float* x0, const int* ei, const int* gidx, float* feats) {
        hipMemcpyAsync(xf, x0, NDc * 4, hipMemcpyDeviceToDevice, stream);
        for (int l = 0; l < Lc; ++l) {
            // agg = segment_sum(x[src], dst)
            hipMemsetAsync(buf1, 0, NDc * 4, stream);
            edge_scatter_kernel<<<(int)(((long)Ec * 16) / TB), TB, 0, stream>>>(xf, ei, buf1, Ec);
            // t16 = f16((1+eps)*x + agg)
            gin_pre_kernel<<<(int)(NDc / TB), TB, 0, stream>>>(xf, buf1, gin_eps, l, t16, NDc);
            // u16 = relu(t16 @ w1 + b1)   [WMMA -> f16]
            gemm64_wmma<true, false, true><<<Nc / 128, 256, 0, stream>>>(
                t16, w16g1 + (size_t)l * 4096, gin_b1 + l * 64, nullptr, u16, 64, 0);
            // h = u16 @ w2 + b2           [WMMA -> f32]
            gemm64_wmma<false, true, false><<<Nc / 128, 256, 0, stream>>>(
                u16, w16g2 + (size_t)l * 4096, gin_b2 + l * 64, buf1, nullptr, 64, 0);
            // batchnorm over all N rows, then relu -> xf (f32) + x16 (f16)
            hipMemsetAsync(stats, 0, 128 * 4, stream);
            bn_stats_kernel<<<Nc / 512, 64, 0, stream>>>(buf1, stats, 512);
            bn_apply_kernel<<<(int)(NDc / TB), TB, 0, stream>>>(buf1, stats,
                bn_gamma + l * 64, bn_beta + l * 64, xf, x16, NDc);
            // inner = relu(x16 @ inner_w + b)  [WMMA -> f32]
            gemm64_wmma<true, true, false><<<Nc / 128, 256, 0, stream>>>(
                x16, w16in + (size_t)l * 4096, inner_b + l * 64, inner, nullptr, 64, 0);
            // pooled = segment_sum(inner, graph_idx)
            hipMemsetAsync(pooled, 0, (size_t)Bc * 64 * 4, stream);
            pool_seg_kernel<<<(int)(NDc / TB), TB, 0, stream>>>(inner, gidx, pooled, NDc);
            cvt_f32_f16_kernel<<<(Bc * 64 + TB - 1) / TB, TB, 0, stream>>>(pooled, pooled16, Bc * 64);
            // feats[:, l*64:(l+1)*64] = relu(pooled @ outer_w + b)  [WMMA -> f32 strided]
            gemm64_wmma<true, true, false><<<Bc / 128, 256, 0, stream>>>(
                pooled16, w16ot + (size_t)l * 4096, outer_b + l * 64, feats, nullptr, 192, l * 64);
        }
    };

    encode(query_x,  qei, qgi, qf);
    encode(corpus_x, cei, cgi, cf);

    // ---- head ----
    diff_kernel<<<(Bc * 192 + TB - 1) / TB, TB, 0, stream>>>(qf, cf, dbuf, Bc * 192);
    mlp_kernel<<<(Bc * 96 + TB - 1) / TB, TB, 0, stream>>>(dbuf, conv_w1, conv_b1, hc1, Bc, 192, 96, 0);
    mlp_kernel<<<(Bc * 96 + TB - 1) / TB, TB, 0, stream>>>(hc1, conv_w2, conv_b2, hc2, Bc, 96, 96, 1);
    score_head_kernel<<<(Bc + TB - 1) / TB, TB, 0, stream>>>(hc2, score_w1, score_b1,
                                                             score_w2, score_b2, scoreb, Bc);
    ntn_kernel<<<(Bc * 16 + TB - 1) / TB, TB, 0, stream>>>(qf, cf, ntn_w, ntn_wb, ntn_bias,
                                                           simrep, Bc);
    sim_head_kernel<<<(Bc + TB - 1) / TB, TB, 0, stream>>>(simrep, sim_w1, sim_b1,
                                                           sim_w2, sim_b2, simb, Bc);
    final_combine_kernel<<<(Bc + TB - 1) / TB, TB, 0, stream>>>(scoreb, simb, alpha, beta,
                                                                (float*)d_out, Bc);
}
